// NeibRoutLayer_37160057045568
// MI455X (gfx1250) — matrix operand order repaired and checked by reading, not compiled, not run on hardware
//
#include <hip/hip_runtime.h>
#include <hip/hip_bf16.h>

typedef __attribute__((ext_vector_type(16))) __bf16 v16bf;
typedef __attribute__((ext_vector_type(8)))  __bf16 v8bf;
typedef __attribute__((ext_vector_type(4)))  __bf16 v4bf;
typedef __attribute__((ext_vector_type(8)))  float  v8f;

#define DD   128
#define KCAP 8
#define NFOLDC 20
#define NITERC 2
#define MT   32      // GEMM M-tile rows per block

// ---------------------------------------------------------------------------
// capnorm: one thread per 16-element capsule group. in==out allowed.
// ---------------------------------------------------------------------------
__global__ void capnorm_kernel(const float* __restrict__ in,
                               float* __restrict__ out, int ngroups) {
  int g = blockIdx.x * blockDim.x + threadIdx.x;
  if (g >= ngroups) return;
  const float4* p = (const float4*)(in + (size_t)g * 16);
  float4 a = p[0], b = p[1], c = p[2], d = p[3];
  float s = a.x*a.x + a.y*a.y + a.z*a.z + a.w*a.w
          + b.x*b.x + b.y*b.y + b.z*b.z + b.w*b.w
          + c.x*c.x + c.y*c.y + c.z*c.z + c.w*c.w
          + d.x*d.x + d.y*d.y + d.z*d.z + d.w*d.w;
  float inv = 1.0f / fmaxf(sqrtf(s), 1e-12f);
  a.x*=inv; a.y*=inv; a.z*=inv; a.w*=inv;
  b.x*=inv; b.y*=inv; b.z*=inv; b.w*=inv;
  c.x*=inv; c.y*=inv; c.z*=inv; c.w*=inv;
  d.x*=inv; d.y*=inv; d.z*=inv; d.w*=inv;
  float4* q = (float4*)(out + (size_t)g * 16);
  q[0]=a; q[1]=b; q[2]=c; q[3]=d;
}

// ---------------------------------------------------------------------------
// One-time: Bcm[col][k] = bf16(x_norm[k][col]), K zero-padded to Kpad.
// Packed 2-element (4B) writes; coalesced reads over col.
// ---------------------------------------------------------------------------
__global__ void xpose_bf16_kernel(const float* __restrict__ X,
                                  __bf16* __restrict__ Bcm, int N_, int Kpad) {
  int i = blockIdx.x * blockDim.x + threadIdx.x;
  int total = (Kpad >> 1) * DD;
  if (i >= total) return;
  int col = i & (DD - 1);
  int k0  = (i >> 7) * 2;
  float v0 = (k0     < N_) ? X[(size_t)k0       * DD + col] : 0.0f;
  float v1 = (k0 + 1 < N_) ? X[(size_t)(k0 + 1) * DD + col] : 0.0f;
  union { __bf16 h[2]; unsigned u; } pk;
  pk.h[0] = (__bf16)v0; pk.h[1] = (__bf16)v1;
  *(unsigned*)(Bcm + (size_t)col * Kpad + k0) = pk.u;
}

// ---------------------------------------------------------------------------
// GEMM C[E x 128] = A[E x N] * B, A f32 converted on the fly, B pre-converted
// column-major bf16 (Bcm[col][Kpad]). Block = 256 threads = 8 waves.
// Block tile: 32 rows x 128 cols; wave w owns cols [16w,16w+16) and issues
// 2 WMMAs per K-step (rows 0-15 and 16-31) sharing one B fragment.
// Branchless staging: clamped addresses + zero masks (no exec divergence
// before the WMMAs; EXEC stays all-ones).
// ---------------------------------------------------------------------------
__global__ void __launch_bounds__(256)
gemm_bf16_wmma(const float* __restrict__ A, const __bf16* __restrict__ Bcm,
               float* __restrict__ C, int E_, int N_, int Kpad) {
  __shared__ __bf16 Abf[MT][40];   // 32 x 32 tile (+pad, 80B row pitch)
  __shared__ __bf16 Bt[DD][40];    // 128 cols x 32 k    (+pad)

  const int tid    = threadIdx.x;
  const int waveId = tid >> 5;
  const int lane   = tid & 31;
  const int half   = lane >> 4;
  const int lid    = lane & 15;
  const int mBase  = blockIdx.x * MT;

  // staging coordinates (fixed per thread)
  const int ar   = tid >> 3;               // A row 0..31
  const int ac   = (tid & 7) * 4;          // A col 0,4,..,28
  const int arow = min(mBase + ar, E_ - 1);      // clamped (dup rows unused)
  const int bcol = tid >> 1;               // B col 0..127
  const int bh   = (tid & 1) * 16;         // B k-half 0 or 16

  v8f acc0 = {}, acc1 = {};

  for (int k0 = 0; k0 < N_; k0 += 32) {
    // ---- stage A (32x32 f32 -> bf16), branchless
    {
      int kk = k0 + ac;
      int kks = min(kk, N_ - 4);           // N_ % 4 == 0, kk % 4 == 0
      float4 v = *(const float4*)(A + (size_t)arow * N_ + kks);
      float zm = (kk < N_) ? 1.0f : 0.0f;  // K-tail zero mask
      v4bf pv;
      pv[0] = (__bf16)(v.x * zm); pv[1] = (__bf16)(v.y * zm);
      pv[2] = (__bf16)(v.z * zm); pv[3] = (__bf16)(v.w * zm);
      *(v4bf*)(&Abf[ar][ac]) = pv;
    }
    // ---- stage B (pure contiguous bf16 copy from Bcm)
    {
      const __bf16* bsrc = Bcm + (size_t)bcol * Kpad + k0 + bh;
      v8bf b0 = ((const v8bf*)bsrc)[0];
      v8bf b1 = ((const v8bf*)bsrc)[1];
      *(v8bf*)(&Bt[bcol][bh])     = b0;
      *(v8bf*)(&Bt[bcol][bh + 8]) = b1;
    }
    // prefetch next A tile (global_prefetch_b8)
    if (k0 + 32 < N_)
      __builtin_prefetch(A + (size_t)arow * N_ + k0 + 32 + ac, 0, 1);
    __syncthreads();

    // ---- fragments per ISA layout (wave32)
    v16bf af0, af1, bfv;
    #pragma unroll
    for (int i = 0; i < 8; ++i) {
      af0[i]     = Abf[lid][half * 8 + i];
      af0[8 + i] = Abf[lid][16 + half * 8 + i];
      af1[i]     = Abf[16 + lid][half * 8 + i];
      af1[8 + i] = Abf[16 + lid][16 + half * 8 + i];
    }
    const int col = waveId * 16 + lid;
    #pragma unroll
    for (int i = 0; i < 16; ++i) bfv[i] = Bt[col][(i < 8) ? (half * 8 + i) : (16 + half * 8 + (i - 8))];

    acc0 = __builtin_amdgcn_wmma_f32_16x16x32_bf16(
        false, af0, false, bfv, (short)0, acc0, false, false);
    acc1 = __builtin_amdgcn_wmma_f32_16x16x32_bf16(
        false, af1, false, bfv, (short)0, acc1, false, false);
    __syncthreads();
  }

  // write back: VGPR j -> row (j + 8*half), col (waveId*16 + lid)
  const int col = waveId * 16 + lid;
  #pragma unroll
  for (int j = 0; j < 8; ++j) {
    int r0 = mBase + j + 8 * half;
    int r1 = r0 + 16;
    if (r0 < E_) C[(size_t)r0 * DD + col] = acc0[j];
    if (r1 < E_) C[(size_t)r1 * DD + col] = acc1[j];
  }
}

// ---------------------------------------------------------------------------
// One routing fold: one wave (32 lanes) per edge; each lane owns 4 of 128
// elements so a capsule group (16 elems) spans 4 lanes. Reads u from the
// pre-fold snapshot, scatter-adds into live u with float atomics.
// ---------------------------------------------------------------------------
__global__ void __launch_bounds__(256)
route_fold(const float* __restrict__ src_feat, const float* __restrict__ u_snap,
           float* __restrict__ u, const int* __restrict__ src_idx,
           const int* __restrict__ dst_idx, int s, int cnt) {
  int wv   = (blockIdx.x * blockDim.x + threadIdx.x) >> 5;
  int lane = threadIdx.x & 31;
  if (wv >= cnt) return;
  int e  = s + wv;
  int si = src_idx[e];
  int di = dst_idx[e];
  const float4 zv = ((const float4*)(src_feat + (size_t)si * DD))[lane];
  const float4 uv = ((const float4*)(u_snap  + (size_t)di * DD))[lane];
  float part = zv.x*uv.x + zv.y*uv.y + zv.z*uv.z + zv.w*uv.w;
  // reduce over the 4 lanes of this capsule group
  part += __shfl_xor(part, 1, 32);
  part += __shfl_xor(part, 2, 32);
  float dg = part;                       // group dot (TAU==1), replicated x4
  // softmax over the 8 groups (each group value replicated on 4 lanes)
  float mx = dg;
  mx = fmaxf(mx, __shfl_xor(mx, 4, 32));
  mx = fmaxf(mx, __shfl_xor(mx, 8, 32));
  mx = fmaxf(mx, __shfl_xor(mx, 16, 32));
  float ex = __expf(dg - mx);
  float sm = ex;
  sm += __shfl_xor(sm, 4, 32);
  sm += __shfl_xor(sm, 8, 32);
  sm += __shfl_xor(sm, 16, 32);          // = sum over the 8 groups
  float p = 4.0f * ex / sm;
  float* uo = u + (size_t)di * DD + 4 * lane;
  atomicAdd(&uo[0], zv.x * p);
  atomicAdd(&uo[1], zv.y * p);
  atomicAdd(&uo[2], zv.z * p);
  atomicAdd(&uo[3], zv.w * p);
}

// ---------------------------------------------------------------------------
// Counting sort of M keys in [0, nbins): histogram -> scan -> scatter
// ---------------------------------------------------------------------------
__global__ void hist_kernel(const int* __restrict__ keys, int* __restrict__ hist,
                            int m) {
  int i = blockIdx.x * blockDim.x + threadIdx.x;
  if (i < m) atomicAdd(&hist[keys[i]], 1);
}

__global__ void __launch_bounds__(1024)
scan_kernel(const int* __restrict__ hist, int* __restrict__ offs, int n) {
  __shared__ int part[1024];
  int t = threadIdx.x;
  int chunk = (n + 1023) / 1024;
  int s = t * chunk;
  int e = min(s + chunk, n);
  int sum = 0;
  for (int i = s; i < e; ++i) sum += hist[i];
  part[t] = sum;
  __syncthreads();
  if (t == 0) {
    int run = 0;
    for (int i = 0; i < 1024; ++i) { int v = part[i]; part[i] = run; run += v; }
  }
  __syncthreads();
  int run = part[t];
  for (int i = s; i < e; ++i) { offs[i] = run; run += hist[i]; }
}

__global__ void scatter_sort(const int* __restrict__ vals,
                             const int* __restrict__ keys,
                             int* __restrict__ cursors,
                             int* __restrict__ vals_out,
                             int* __restrict__ keys_out, int m) {
  int i = blockIdx.x * blockDim.x + threadIdx.x;
  if (i >= m) return;
  int k = keys[i];
  int pos = atomicAdd(&cursors[k], 1);
  keys_out[pos] = k;
  vals_out[pos] = vals[i];
}

// ---------------------------------------------------------------------------
extern "C" void kernel_launch(void* const* d_in, const int* in_sizes, int n_in,
                              void* d_out, int out_size, void* d_ws, size_t ws_size,
                              hipStream_t stream) {
  const float* x    = (const float*)d_in[0];
  const float* adj  = (const float*)d_in[1];
  const int*   enm  = (const int*)d_in[2];

  const int N = in_sizes[0] / DD;
  const int M = in_sizes[2] / 2;
  const int E = in_sizes[1] / N;
  const int fold_len = M / NFOLDC;
  const int Kpad = (N + 31) & ~31;

  const int* edge_es = enm;        // edge indices [0,E)
  const int* node_es = enm + M;    // node indices [0,N)

  // workspace carve-up (all 16B-aligned: DD=128 element rows)
  float*  x_norm = (float*)d_ws;               // N*DD
  float*  u_edge = x_norm + (size_t)N * DD;    // E*DD
  float*  u_node = u_edge + (size_t)E * DD;    // N*DD
  float*  u_snap = u_node + (size_t)N * DD;    // max(N,E)*DD = N*DD
  __bf16* Bcm    = (__bf16*)(u_snap + (size_t)N * DD);   // DD*Kpad bf16
  int*    hist    = (int*)(Bcm + (size_t)DD * Kpad);
  int*    offs    = hist + N;
  int*    cursors = offs + N;
  int*    edge_ns = cursors + N;               // M
  int*    node_ns = edge_ns + M;               // M

  // 1. x_norm = capnorm(x)
  {
    int ng = N * KCAP;
    capnorm_kernel<<<(ng + 255) / 256, 256, 0, stream>>>(x, x_norm, ng);
  }

  // 2a. Bcm = transpose+bf16(x_norm), zero-padded K
  {
    int tot = (Kpad / 2) * DD;
    xpose_bf16_kernel<<<(tot + 255) / 256, 256, 0, stream>>>(x_norm, Bcm, N, Kpad);
  }

  // 2b. u_edge = adj @ x_norm  (bf16 WMMA, f32 accumulate)
  {
    int mtiles = (E + MT - 1) / MT;
    gemm_bf16_wmma<<<mtiles, 256, 0, stream>>>(adj, Bcm, u_edge, E, N, Kpad);
  }

  // 3. u_edge = capnorm(u_edge)
  {
    int ng = E * KCAP;
    capnorm_kernel<<<(ng + 255) / 256, 256, 0, stream>>>(u_edge, u_edge, ng);
  }

  // 4. routing 1: src=x_norm, u=u_edge, src_idx=node_es, dst_idx=edge_es
  for (int it = 0; it < NITERC; ++it) {
    for (int f = 0; f < NFOLDC; ++f) {
      int s = f * fold_len;
      int t = (f == NFOLDC - 1) ? M : (f + 1) * fold_len;
      int cnt = t - s;
      hipMemcpyAsync(u_snap, u_edge, (size_t)E * DD * sizeof(float),
                     hipMemcpyDeviceToDevice, stream);
      route_fold<<<(cnt + 7) / 8, 256, 0, stream>>>(
          x_norm, u_snap, u_edge, node_es, edge_es, s, cnt);
    }
    int ng = E * KCAP;
    capnorm_kernel<<<(ng + 255) / 256, 256, 0, stream>>>(u_edge, u_edge, ng);
  }

  // 5. counting sort of node_es
  hipMemsetAsync(hist, 0, (size_t)N * sizeof(int), stream);
  hist_kernel<<<(M + 255) / 256, 256, 0, stream>>>(node_es, hist, M);
  scan_kernel<<<1, 1024, 0, stream>>>(hist, offs, N);
  hipMemcpyAsync(cursors, offs, (size_t)N * sizeof(int),
                 hipMemcpyDeviceToDevice, stream);
  scatter_sort<<<(M + 255) / 256, 256, 0, stream>>>(edge_es, node_es, cursors,
                                                    edge_ns, node_ns, M);

  // 6. u_node starts as capnormed x
  hipMemcpyAsync(u_node, x_norm, (size_t)N * DD * sizeof(float),
                 hipMemcpyDeviceToDevice, stream);

  // 7. routing 2: src=u_edge, u=u_node, src_idx=edge_ns, dst_idx=node_ns
  for (int it = 0; it < NITERC; ++it) {
    for (int f = 0; f < NFOLDC; ++f) {
      int s = f * fold_len;
      int t = (f == NFOLDC - 1) ? M : (f + 1) * fold_len;
      int cnt = t - s;
      hipMemcpyAsync(u_snap, u_node, (size_t)N * DD * sizeof(float),
                     hipMemcpyDeviceToDevice, stream);
      route_fold<<<(cnt + 7) / 8, 256, 0, stream>>>(
          u_edge, u_snap, u_node, edge_ns, node_ns, s, cnt);
    }
    int ng = N * KCAP;
    capnorm_kernel<<<(ng + 255) / 256, 256, 0, stream>>>(u_node, u_node, ng);
  }

  // 8. outputs: [u_node_all (N*DD), u_edge_all (E*DD)]
  hipMemcpyAsync(d_out, u_node, (size_t)N * DD * sizeof(float),
                 hipMemcpyDeviceToDevice, stream);
  hipMemcpyAsync((float*)d_out + (size_t)N * DD, u_edge,
                 (size_t)E * DD * sizeof(float),
                 hipMemcpyDeviceToDevice, stream);
}
